// Attention_regular_27032524161141
// MI455X (gfx1250) — matrix operand order repaired and checked by reading, not compiled, not run on hardware
//
#include <hip/hip_runtime.h>
#include <hip/hip_bf16.h>

typedef __attribute__((ext_vector_type(16))) _Float16 v16h;
typedef __attribute__((ext_vector_type(4)))  _Float16 h4;
typedef __attribute__((ext_vector_type(8)))  float    v8f;

#define HEADS 6
#define HD    32
#define NWIN  128   // windows per image (16 x 8)
#define NTOK  256   // tokens per window (8 x 32)
#define BSZ   4
#define HIMG  128
#define WIMG  256
#define CDIM  192
#define LIMG  (HIMG * WIMG)
#define PD    12
#define NBIAS 945

// ---------------------------------------------------------------------------
// Kernel 1: DynamicPosBias MLP: (945,2) -> (945,6)
// ---------------------------------------------------------------------------
__global__ void pos_mlp_kernel(const float* __restrict__ rpe,
                               const float* __restrict__ pw0, const float* __restrict__ pb0,
                               const float* __restrict__ g1, const float* __restrict__ be1,
                               const float* __restrict__ w1, const float* __restrict__ b1,
                               const float* __restrict__ g2, const float* __restrict__ be2,
                               const float* __restrict__ w2, const float* __restrict__ b2,
                               const float* __restrict__ g3, const float* __restrict__ be3,
                               const float* __restrict__ w3, const float* __restrict__ b3,
                               float* __restrict__ pos) {
  int r = blockIdx.x * blockDim.x + threadIdx.x;
  if (r >= NBIAS) return;
  float x[PD], y[PD];
  float r0 = rpe[r * 2 + 0], r1 = rpe[r * 2 + 1];
  #pragma unroll
  for (int j = 0; j < PD; ++j)
    x[j] = pw0[j * 2 + 0] * r0 + pw0[j * 2 + 1] * r1 + pb0[j];

  const float* G[3]  = {g1, g2, g3};
  const float* BE[3] = {be1, be2, be3};
  const float* Wm[3] = {w1, w2, w3};
  const float* Bb[3] = {b1, b2, b3};
  #pragma unroll
  for (int layer = 0; layer < 3; ++layer) {
    float m = 0.f;
    #pragma unroll
    for (int j = 0; j < PD; ++j) m += x[j];
    m *= (1.0f / PD);
    float v = 0.f;
    #pragma unroll
    for (int j = 0; j < PD; ++j) { float d = x[j] - m; v += d * d; }
    v *= (1.0f / PD);
    float is = rsqrtf(v + 1e-5f);
    #pragma unroll
    for (int j = 0; j < PD; ++j) {
      float t = (x[j] - m) * is * G[layer][j] + BE[layer][j];
      y[j] = t > 0.f ? t : 0.f;
    }
    if (layer < 2) {
      #pragma unroll
      for (int o = 0; o < PD; ++o) {
        float a = Bb[layer][o];
        #pragma unroll
        for (int j = 0; j < PD; ++j) a += Wm[layer][o * PD + j] * y[j];
        x[o] = a;
      }
    } else {
      #pragma unroll
      for (int h = 0; h < HEADS; ++h) {
        float a = Bb[layer][h];
        #pragma unroll
        for (int j = 0; j < PD; ++j) a += Wm[layer][h * PD + j] * y[j];
        pos[r * HEADS + h] = a;
      }
    }
  }
}

// ---------------------------------------------------------------------------
// Kernel 2: rpb[h][n*256+m] = pos[rpi[n*256+m]][h]
// ---------------------------------------------------------------------------
__global__ void rpb_gather_kernel(const int* __restrict__ rpi,
                                  const float* __restrict__ pos,
                                  float* __restrict__ rpb) {
  int id = blockIdx.x * blockDim.x + threadIdx.x;   // 0..65535
  int r = rpi[id];
  #pragma unroll
  for (int h = 0; h < HEADS; ++h)
    rpb[h * (NTOK * NTOK) + id] = pos[r * HEADS + h];
}

// ---------------------------------------------------------------------------
// Kernel 3: fused windowed attention, one block per (window, head)
// ---------------------------------------------------------------------------
__global__ __launch_bounds__(256) void attn_kernel(const float* __restrict__ qkv,
                                                   const float* __restrict__ mask,
                                                   const float* __restrict__ rpb,
                                                   float* __restrict__ out) {
  __shared__ __align__(16) _Float16 Qh[NTOK * HD];   // [query][dim]  16 KB
  __shared__ __align__(16) _Float16 Kh[NTOK * HD];   // [key][dim]    16 KB
  __shared__ __align__(16) _Float16 Vt[HD * NTOK];   // [dim][key]    16 KB
  __shared__ __align__(16) _Float16 Pbuf[8][16 * 32];// per-wave P     8 KB

  const int h  = blockIdx.x;          // head
  const int w  = blockIdx.y;          // global window 0..511
  const int b  = w / NWIN;
  const int wi = w % NWIN;
  const int wy = wi >> 3;             // 0..15
  const int wx = wi & 7;              // 0..7
  const int tid  = threadIdx.x;
  const int lane = tid & 31;
  const int wave = tid >> 5;
  const int half = lane >> 4;         // 0 or 1
  const int ln   = lane & 15;

  // ---- stage Q (scaled) / K / V(transposed) into LDS as f16 ----
  {
    const int n = tid;                         // token in window
    const int y = n >> 5, x = n & 31;
    const long l = (long)(wy * 8 + y) * WIMG + (wx * 32 + x);
    const long base = ((long)b * LIMG + l) * CDIM + h * HD;
    const float4* q4 = (const float4*)(qkv + base);
    const float4* k4 = (const float4*)(qkv + (long)BSZ * LIMG * CDIM + base);
    const float4* v4 = (const float4*)(qkv + 2l * (long)BSZ * LIMG * CDIM + base);
    const float scale = 0.17677669529663687f;  // 32^-0.5
    #pragma unroll
    for (int dd = 0; dd < 8; ++dd) {
      float4 qv = q4[dd];
      float4 kv = k4[dd];
      float4 vv = v4[dd];
      h4 qh = {(_Float16)(qv.x * scale), (_Float16)(qv.y * scale),
               (_Float16)(qv.z * scale), (_Float16)(qv.w * scale)};
      h4 kh = {(_Float16)kv.x, (_Float16)kv.y, (_Float16)kv.z, (_Float16)kv.w};
      *(h4*)&Qh[n * HD + dd * 4] = qh;
      *(h4*)&Kh[n * HD + dd * 4] = kh;
      Vt[(dd * 4 + 0) * NTOK + n] = (_Float16)vv.x;
      Vt[(dd * 4 + 1) * NTOK + n] = (_Float16)vv.y;
      Vt[(dd * 4 + 2) * NTOK + n] = (_Float16)vv.z;
      Vt[(dd * 4 + 3) * NTOK + n] = (_Float16)vv.w;
    }
  }
  __syncthreads();

  const float* mrow = mask + (long)wi * (NTOK * NTOK);
  const float* rrow = rpb + (long)h * (NTOK * NTOK);

  for (int rbi = 0; rbi < 2; ++rbi) {
    const int rb = wave + rbi * 8;            // query row-block 0..15

    // per-lane row offsets for bias loads (fixed for this row-block)
    int qoff[8];
    #pragma unroll
    for (int i = 0; i < 8; ++i)
      qoff[i] = (rb * 16 + i + half * 8) * NTOK;

    // A fragment of Q (16x32, interleaved K-chunk layout)
    v16h qa;
    {
      const _Float16* p = Qh + (rb * 16 + ln) * HD + half * 8;
      *((uint4*)&qa)     = *(const uint4*)p;          // K chunk 0 (8 halfs)
      *((uint4*)&qa + 1) = *(const uint4*)(p + 16);   // K chunk 1
    }

    // ---- S = Q K^T + (mask + rpb) for all 16 key blocks ----
    // bias tile goes in as the WMMA C operand: D = A*B + C
    v8f s[16];
    #pragma unroll
    for (int j = 0; j < 16; ++j) {
      const int kg = j * 16 + ln;
      float mv[8], rv[8];
      #pragma unroll
      for (int i = 0; i < 8; ++i) mv[i] = mrow[qoff[i] + kg];
      #pragma unroll
      for (int i = 0; i < 8; ++i) rv[i] = rrow[qoff[i] + kg];

      v16h kb;  // B fragment: lane ln holds key j*16+ln, interleaved K-chunk dims
      {
        const _Float16* p = Kh + (j * 16 + ln) * HD + half * 8;
        *((uint4*)&kb)     = *(const uint4*)p;
        *((uint4*)&kb + 1) = *(const uint4*)(p + 16);
      }
      v8f c;
      #pragma unroll
      for (int i = 0; i < 8; ++i) c[i] = mv[i] + rv[i];
      s[j] = __builtin_amdgcn_wmma_f32_16x16x32_f16(false, qa, false, kb,
                                                    (short)0, c, false, false);
    }

    // ---- row max (over keys: registers + 16-lane butterfly) ----
    float rmax[8], rsum[8];
    #pragma unroll
    for (int i = 0; i < 8; ++i) {
      float m = s[0][i];
      #pragma unroll
      for (int j = 1; j < 16; ++j) m = fmaxf(m, s[j][i]);
      m = fmaxf(m, __shfl_xor(m, 1));
      m = fmaxf(m, __shfl_xor(m, 2));
      m = fmaxf(m, __shfl_xor(m, 4));
      m = fmaxf(m, __shfl_xor(m, 8));
      rmax[i] = m;
    }
    // ---- exp + row sum ----
    #pragma unroll
    for (int i = 0; i < 8; ++i) {
      float t = 0.f;
      #pragma unroll
      for (int j = 0; j < 16; ++j) {
        float e = __expf(s[j][i] - rmax[i]);
        s[j][i] = e;
        t += e;
      }
      t += __shfl_xor(t, 1);
      t += __shfl_xor(t, 2);
      t += __shfl_xor(t, 4);
      t += __shfl_xor(t, 8);
      rsum[i] = t;
    }

    // ---- O = P V  (8 chunks of 32 keys; P bounced through per-wave LDS) ----
    v8f o0 = {}, o1 = {};
    _Float16* pb = &Pbuf[wave][0];
    #pragma unroll
    for (int cch = 0; cch < 8; ++cch) {
      #pragma unroll
      for (int jj = 0; jj < 2; ++jj) {
        v8f t = s[cch * 2 + jj];
        #pragma unroll
        for (int i = 0; i < 8; ++i)
          pb[(i + half * 8) * 32 + jj * 16 + ln] = (_Float16)t[i];
      }
      v16h pa;  // A fragment of P (16 queries x 32 keys)
      {
        const _Float16* p = pb + ln * 32 + half * 8;
        *((uint4*)&pa)     = *(const uint4*)p;
        *((uint4*)&pa + 1) = *(const uint4*)(p + 16);
      }
      v16h vb0, vb1;  // B fragments of V: dims 0..15 and 16..31
      {
        const _Float16* p = Vt + ln * NTOK + cch * 32 + half * 8;
        *((uint4*)&vb0)     = *(const uint4*)p;
        *((uint4*)&vb0 + 1) = *(const uint4*)(p + 16);
        const _Float16* p1 = p + 16 * NTOK;
        *((uint4*)&vb1)     = *(const uint4*)p1;
        *((uint4*)&vb1 + 1) = *(const uint4*)(p1 + 16);
      }
      o0 = __builtin_amdgcn_wmma_f32_16x16x32_f16(false, pa, false, vb0,
                                                  (short)0, o0, false, false);
      o1 = __builtin_amdgcn_wmma_f32_16x16x32_f16(false, pa, false, vb1,
                                                  (short)0, o1, false, false);
    }

    // ---- normalize and write out (B,H,W,C) ----
    #pragma unroll
    for (int i = 0; i < 8; ++i) {
      const int n = rb * 16 + i + half * 8;   // query token in window
      const int y = n >> 5, x = n & 31;
      const long l = (long)(wy * 8 + y) * WIMG + (wx * 32 + x);
      float* op = out + ((long)b * LIMG + l) * CDIM + h * HD;
      const float inv = 1.0f / rsum[i];
      op[ln]      = o0[i] * inv;
      op[16 + ln] = o1[i] * inv;
    }
  }
}

// ---------------------------------------------------------------------------
extern "C" void kernel_launch(void* const* d_in, const int* in_sizes, int n_in,
                              void* d_out, int out_size, void* d_ws, size_t ws_size,
                              hipStream_t stream) {
  const float* qkv  = (const float*)d_in[0];
  const float* mask = (const float*)d_in[1];
  const float* rpe  = (const float*)d_in[2];
  const float* pw0  = (const float*)d_in[3];
  const float* pb0  = (const float*)d_in[4];
  const float* g1   = (const float*)d_in[5];
  const float* be1  = (const float*)d_in[6];
  const float* w1   = (const float*)d_in[7];
  const float* b1   = (const float*)d_in[8];
  const float* g2   = (const float*)d_in[9];
  const float* be2  = (const float*)d_in[10];
  const float* w2   = (const float*)d_in[11];
  const float* b2   = (const float*)d_in[12];
  const float* g3   = (const float*)d_in[13];
  const float* be3  = (const float*)d_in[14];
  const float* w3   = (const float*)d_in[15];
  const float* b3   = (const float*)d_in[16];
  const int*   rpi  = (const int*)d_in[17];
  float* out = (float*)d_out;

  float* pos = (float*)d_ws;                       // 945*6 floats
  float* rpb = (float*)((char*)d_ws + 32768);      // 6*256*256 floats

  pos_mlp_kernel<<<4, 256, 0, stream>>>(rpe, pw0, pb0, g1, be1, w1, b1,
                                        g2, be2, w2, b2, g3, be3, w3, b3, pos);
  rpb_gather_kernel<<<256, 256, 0, stream>>>(rpi, pos, rpb);

  dim3 grid(HEADS, BSZ * NWIN);   // (6, 512)
  attn_kernel<<<grid, 256, 0, stream>>>(qkv, mask, rpb, out);
}
